// DifuscoBackbone_10728828305686
// MI455X (gfx1250) — matrix operand order repaired and checked by reading, not compiled, not run on hardware
//
#include <hip/hip_runtime.h>

// DifuscoBackbone on MI455X (gfx1250): fused per-layer edge pipeline with
// f16 WMMA (f32 accum). Memory-bound on the E x D edge tensor, so each layer
// is ONE pass over e: GEMM(e@P + gathered HQ/HR) -> LN -> gate/scatter ->
// MLP(2x GEMM) -> residual, all with e_hat resident in WMMA accumulators.
// Data movement: TDM tensor_load_to_lds for activation tiles; double-buffered
// global_load_async_to_lds_b128 weight panels overlapped with WMMA via
// s_wait_asynccnt <= 4 (4 in-flight b128 per thread per panel).

#define D   256
#define LAY 12
#define NN  10000
#define EE  320000

typedef __attribute__((ext_vector_type(16))) _Float16 v16h;
typedef __attribute__((ext_vector_type(8)))  float    v8f;
typedef __attribute__((ext_vector_type(4)))  unsigned u32x4;
typedef __attribute__((ext_vector_type(8)))  int      i32x8;
typedef __attribute__((ext_vector_type(4)))  int      i32x4;

#if defined(__has_builtin)
#  if __has_builtin(__builtin_amdgcn_tensor_load_to_lds)
#    define HAVE_TDM 1
#  endif
#  if __has_builtin(__builtin_amdgcn_s_wait_asynccnt)
#    define HAVE_WAIT_ASYNC_BUILTIN 1
#  endif
#endif
#ifndef HAVE_TDM
#  define HAVE_TDM 0
#endif
#ifndef HAVE_WAIT_ASYNC_BUILTIN
#  define HAVE_WAIT_ASYNC_BUILTIN 0
#endif

__device__ __forceinline__ v8f wmma_f32_16x16x32(v16h a, v16h b, v8f c) {
  // (neg_a, A, neg_b, B, c_mod, C, reuse_a, reuse_b)
  return __builtin_amdgcn_wmma_f32_16x16x32_f16(false, a, false, b, (short)0, c,
                                                false, false);
}

// A fragment: 16x32 f16 tile from a row-major LDS slab (ld = D halves).
__device__ __forceinline__ v16h fragA(const _Float16* base, int lane, int k0) {
  int row = lane & 15;
  int ko  = k0 + ((lane >> 4) << 4);
  return *(const v16h*)(base + row * D + ko);
}

// B fragment: 32x16 f16 tile from a 32 x D row-major LDS weight panel.
__device__ __forceinline__ v16h fragB(const _Float16* bpanel, int lane, int col0) {
  const _Float16* p = bpanel + ((lane >> 4) << 4) * D + col0 + (lane & 15);
  v16h b;
#pragma unroll
  for (int s = 0; s < 16; ++s) b[s] = p[s * D];
  return b;
}

// ---------------------------------------------------------------------------
// CDNA5 async data movement
// ---------------------------------------------------------------------------
// Direct memory->LDS copy, bypassing VGPRs. On amdgcn the low 32 bits of a
// generic pointer into LDS are the LDS byte offset.
__device__ __forceinline__ void async_b128(_Float16* ldst, const _Float16* gsrc) {
  unsigned           l = (unsigned)(unsigned long long)ldst;
  unsigned long long g = (unsigned long long)gsrc;
  asm volatile("global_load_async_to_lds_b128 %0, %1, off"
               :: "v"(l), "v"(g) : "memory");
}

__device__ __forceinline__ void wait_async0() {
#if HAVE_WAIT_ASYNC_BUILTIN
  __builtin_amdgcn_s_wait_asynccnt(0);
#else
  asm volatile("s_wait_asynccnt 0x0" ::: "memory");
#endif
}

// Allow 4 newer async loads (next panel) to remain in flight; older panel's
// loads (issued earlier, completing in order) are guaranteed done.
__device__ __forceinline__ void wait_async4() {
#if HAVE_WAIT_ASYNC_BUILTIN
  __builtin_amdgcn_s_wait_asynccnt(4);
#else
  asm volatile("s_wait_asynccnt 0x4" ::: "memory");
#endif
}

// Issue (no wait) a 32 x D f16 weight panel into LDS: 4 b128 per thread.
__device__ __forceinline__ void stage_nowait(_Float16* sB, const _Float16* W,
                                             int tid) {
  for (int c = tid; c < (32 * D) >> 3; c += 256)
    async_b128(sB + (c << 3), W + (c << 3));
}

#if HAVE_TDM
// TDM: DMA a rows x D f16 tile (row-major, stride D) from global into LDS.
// tensorRows bounds dim1 so OOB rows zero-fill (handles ragged tiles).
__device__ __forceinline__ void tdm_load_tile(_Float16* ldst, const _Float16* gsrc,
                                              unsigned rows, unsigned tensorRows) {
  unsigned long long ga = (unsigned long long)gsrc;
  unsigned lds = (unsigned)(unsigned long long)ldst;
  u32x4 g0;
  g0[0] = 1u;                                              // count=1 user D#
  g0[1] = lds;                                             // lds_addr
  g0[2] = (unsigned)(ga & 0xFFFFFFFFu);                    // global_addr lo
  g0[3] = (unsigned)((ga >> 32) & 0x1FFFFFFu) | (2u << 30);// addr hi | type=2
  i32x8 g1;
  g1[0] = (int)(1u << 16);                                 // data_size=2B
  g1[1] = (int)((D & 0xFFFFu) << 16);                      // tensor_dim0 lo16
  g1[2] = (int)(((unsigned)D >> 16) & 0xFFFFu) |
          (int)((tensorRows & 0xFFFFu) << 16);             // dim0 hi | dim1 lo
  g1[3] = (int)((tensorRows >> 16) & 0xFFFFu) |
          (int)((unsigned)D << 16);                        // dim1 hi | tile_dim0
  g1[4] = (int)(rows & 0xFFFFu);                           // tile_dim1 (dim2=0)
  g1[5] = D;                                               // dim0_stride lo32
  g1[6] = 0;
  g1[7] = 0;
  i32x4 z4 = {0, 0, 0, 0};
#if __clang_major__ >= 23
  i32x8 z8 = {0, 0, 0, 0, 0, 0, 0, 0};
  __builtin_amdgcn_tensor_load_to_lds(g0, g1, z4, z4, z8, 0);
#else
  __builtin_amdgcn_tensor_load_to_lds(g0, g1, z4, z4, 0);
#endif
  __builtin_amdgcn_s_wait_tensorcnt(0);
}
#endif

// Stage a 64 x D activation tile (rows clamped/zero-filled against M).
__device__ __forceinline__ void stage_tile64(_Float16* sT, const _Float16* A,
                                             int r0, int M, int tid, int wave) {
#if HAVE_TDM
  if (wave == 0)
    tdm_load_tile(sT, A + (size_t)r0 * D, 64u, (unsigned)M);
  (void)tid;
#else
  for (int c = tid; c < (64 * D) >> 3; c += 256) {
    int row = c >> 5;
    int gr  = r0 + row; if (gr >= M) gr = M - 1;
    *(float4*)(sT + (c << 3)) =
        *(const float4*)(A + (size_t)gr * D + ((c & 31) << 3));
  }
  (void)wave;
#endif
}

// ---------------------------------------------------------------------------
// One full K-pass: acc[8] += A_lds(16 x D slab at aBase) @ W(D x D), with the
// 32 x D weight panel double-buffered through async-to-LDS.
//   sB2 must hold 2 * 32 * D halves. Must be entered with block in sync.
// ---------------------------------------------------------------------------
__device__ __forceinline__ void gemm_pass(v8f (&acc)[8], const _Float16* aBase,
                                          const _Float16* W, _Float16* sB2,
                                          int tid, int lane, int colBase) {
  stage_nowait(sB2, W, tid);  // panel 0 in flight
  for (int k0 = 0; k0 < D; k0 += 32) {
    _Float16* cur = sB2 + ((k0 >> 5) & 1) * (32 * D);
    if (k0 + 32 < D) {
      _Float16* nxt = sB2 + (((k0 >> 5) + 1) & 1) * (32 * D);
      stage_nowait(nxt, W + (size_t)(k0 + 32) * D, tid);
      if (k0 + 64 < D) __builtin_prefetch(W + (size_t)(k0 + 64) * D, 0, 1);
      wait_async4();          // older panel done; next panel still in flight
    } else {
      wait_async0();
    }
    __syncthreads();
    v16h a = fragA(aBase, lane, k0);
#pragma unroll
    for (int j = 0; j < 8; ++j)
      acc[j] = wmma_f32_16x16x32(a, fragB(cur, lane, colBase + j * 16), acc[j]);
    __syncthreads();
  }
}

// ---------------------------------------------------------------------------
// Small elementwise / embedding kernels
// ---------------------------------------------------------------------------
__global__ void k_cvt(const float* __restrict__ s, _Float16* __restrict__ d, int n) {
  int t = blockIdx.x * blockDim.x + threadIdx.x;
  if (t < n) d[t] = (_Float16)s[t];
}

__global__ void k_zero(float* __restrict__ p, int n) {
  int t = blockIdx.x * blockDim.x + threadIdx.x;
  if (t < n) p[t] = 0.0f;
}

__global__ void k_embed_nodes(const float* __restrict__ coords,
                              float* __restrict__ h32, _Float16* __restrict__ h16) {
  int t = blockIdx.x * blockDim.x + threadIdx.x;
  if (t >= NN * D) return;
  int i = t >> 8, c = t & 255;
  float v  = coords[i * 2 + (c >> 7)];
  int   cc = c & 127;
  int   j  = cc & 63;
  float f  = __expf(-logf(10000.0f) * (float)j * (1.0f / 64.0f));
  float a  = v * f;
  float val = (cc < 64) ? __sinf(a) : __cosf(a);
  h32[t] = val;
  h16[t] = (_Float16)val;
}

__global__ void k_embed_edges(const float* __restrict__ dist,
                              const float* __restrict__ xt,
                              _Float16* __restrict__ e16) {
  long long t = (long long)blockIdx.x * blockDim.x + threadIdx.x;
  if (t >= (long long)EE * D) return;
  int i = (int)(t >> 8), c = (int)(t & 255);
  float v  = (c < 128) ? dist[i] : xt[i];
  int   cc = c & 127;
  int   j  = cc & 63;
  float f  = __expf(-logf(10000.0f) * (float)j * (1.0f / 64.0f));
  float a  = v * f;
  float val = (cc < 64) ? __sinf(a) : __cosf(a);
  e16[t] = (_Float16)val;
}

__global__ __launch_bounds__(256) void k_time_embed(
    const float* __restrict__ t, const float* __restrict__ W1,
    const float* __restrict__ B1, const float* __restrict__ W2,
    const float* __restrict__ B2, float* __restrict__ te) {
  __shared__ float s0[D];
  __shared__ float s1[D];
  int c = threadIdx.x;
  float tv = t[0];
  int   j  = c & 127;
  float f  = __expf(-logf(10000.0f) * (float)j * (1.0f / 128.0f));
  float a  = tv * f;
  s0[c] = (c < 128) ? __sinf(a) : __cosf(a);
  __syncthreads();
  float acc = B1[c];
  for (int k = 0; k < D; ++k) acc += s0[k] * W1[k * D + c];
  acc = acc / (1.0f + __expf(-acc));     // silu
  s1[c] = acc;
  __syncthreads();
  float acc2 = B2[c];
  for (int k = 0; k < D; ++k) acc2 += s1[k] * W2[k * D + c];
  te[c] = acc2;
}

__global__ __launch_bounds__(256) void k_mlp_t(
    const float* __restrict__ te, const float* __restrict__ W1,
    const float* __restrict__ B1, const float* __restrict__ W2,
    const float* __restrict__ B2, float* __restrict__ out) {
  __shared__ float s1[D];
  int c = threadIdx.x;
  float acc = B1[c];
  for (int k = 0; k < D; ++k) acc += te[k] * W1[k * D + c];
  s1[c] = fmaxf(acc, 0.0f);
  __syncthreads();
  float acc2 = B2[c];
  for (int k = 0; k < D; ++k) acc2 += s1[k] * W2[k * D + c];
  out[c] = acc2;
}

// ---------------------------------------------------------------------------
// Generic WMMA GEMM: C[M x D] = A[M x D] @ W[D x D], f16 in, f32 accum.
// ---------------------------------------------------------------------------
__global__ __launch_bounds__(256) void k_gemm256(
    const _Float16* __restrict__ A, const _Float16* __restrict__ W,
    _Float16* __restrict__ Ch, float* __restrict__ Cf, int M) {
  __shared__ alignas(32) _Float16 sA[64 * D];
  __shared__ alignas(32) _Float16 sB2[2 * 32 * D];
  int tid = threadIdx.x, lane = tid & 31, wave = tid >> 5;
  int r0 = blockIdx.x * 64;
  int rbase   = (wave & 3) * 16;
  int colBase = (wave >> 2) * 128;
  int rl = (lane >> 4) << 3;
  int cl = lane & 15;

  stage_tile64(sA, A, r0, M, tid, wave);
  __syncthreads();

  v8f acc[8] = {};
  gemm_pass(acc, sA + rbase * D, W, sB2, tid, lane, colBase);

#pragma unroll
  for (int j = 0; j < 8; ++j)
#pragma unroll
    for (int r = 0; r < 8; ++r) {
      int row = r0 + rbase + rl + r;
      int col = colBase + j * 16 + cl;
      if (row < M) {
        if (Ch) Ch[(size_t)row * D + col] = (_Float16)acc[j][r];
        if (Cf) Cf[(size_t)row * D + col] = acc[j][r];
      }
    }
}

// ---------------------------------------------------------------------------
// Fused per-layer edge kernel. Block tile = 64 edges x D.
// ---------------------------------------------------------------------------
__global__ __launch_bounds__(256) void k_edge_layer(
    _Float16* __restrict__ e16,
    const int* __restrict__ srcIdx, const int* __restrict__ dstIdx,
    const _Float16* __restrict__ HQ, const _Float16* __restrict__ HR,
    const _Float16* __restrict__ HV,
    const _Float16* __restrict__ Pw, const _Float16* __restrict__ W1,
    const _Float16* __restrict__ W2,
    const float* __restrict__ lnS, const float* __restrict__ lnB,
    const float* __restrict__ b1, const float* __restrict__ b2,
    const float* __restrict__ mlpT, float* __restrict__ agg) {
  __shared__ alignas(32) _Float16 sT[64 * D];     // e tile, reused as z tile
  __shared__ alignas(32) _Float16 sEN[64 * D];    // normalized activations
  __shared__ alignas(32) _Float16 sB2[2 * 32 * D];// double-buffered panel
  __shared__ float sSum[64], sSsq[64];
  __shared__ int   sS[64], sD[64];

  int tid = threadIdx.x, lane = tid & 31, wave = tid >> 5;
  int r0 = blockIdx.x * 64;
  int rbase   = (wave & 3) * 16;
  int colBase = (wave >> 2) * 128;
  int rl = (lane >> 4) << 3;
  int cl = lane & 15;

  // Stage e tile (TDM) + indices.
  stage_tile64(sT, e16, r0, EE, tid, wave);
  if (tid < 64) {
    sS[tid] = srcIdx[r0 + tid];
    sD[tid] = dstIdx[r0 + tid];
    sSum[tid] = 0.0f; sSsq[tid] = 0.0f;
  }
  __syncthreads();

  // acc <- HQ[src] + HR[dst]  (gathered)
  v8f acc[8];
#pragma unroll
  for (int j = 0; j < 8; ++j)
#pragma unroll
    for (int r = 0; r < 8; ++r) {
      int row = rbase + rl + r;
      int col = colBase + j * 16 + cl;
      acc[j][r] = (float)HQ[(size_t)sS[row] * D + col] +
                  (float)HR[(size_t)sD[row] * D + col];
    }

  // acc += e @ P
  gemm_pass(acc, sT + rbase * D, Pw, sB2, tid, lane, colBase);

  // Row mean / var via LDS float atomics.
#pragma unroll
  for (int r = 0; r < 8; ++r) {
    float ps = 0.0f, pq = 0.0f;
#pragma unroll
    for (int j = 0; j < 8; ++j) { float v = acc[j][r]; ps += v; pq += v * v; }
    atomicAdd(&sSum[rbase + rl + r], ps);
    atomicAdd(&sSsq[rbase + rl + r], pq);
  }
  __syncthreads();

  // Normalize -> sEN; gate * HV[dst] scattered into agg[src].
#pragma unroll
  for (int r = 0; r < 8; ++r) {
    int row  = rbase + rl + r;
    float mu   = sSum[row] * (1.0f / D);
    float var  = sSsq[row] * (1.0f / D) - mu * mu;
    float rstd = rsqrtf(var + 1e-5f);
    int s = sS[row], d = sD[row];
#pragma unroll
    for (int j = 0; j < 8; ++j) {
      int col  = colBase + j * 16 + cl;
      float eh = acc[j][r];
      float en = (eh - mu) * rstd * lnS[col] + lnB[col];
      sEN[row * D + col] = (_Float16)en;
      float gate = 1.0f / (1.0f + __expf(-eh));
      atomicAdd(&agg[(size_t)s * D + col], gate * (float)HV[(size_t)d * D + col]);
    }
  }
  __syncthreads();

  // z = relu(en @ W1 + b1)  -> reuse sT
#pragma unroll
  for (int j = 0; j < 8; ++j) {
    float bv = b1[colBase + j * 16 + cl];
#pragma unroll
    for (int r = 0; r < 8; ++r) acc[j][r] = bv;
  }
  gemm_pass(acc, sEN + rbase * D, W1, sB2, tid, lane, colBase);
#pragma unroll
  for (int r = 0; r < 8; ++r)
#pragma unroll
    for (int j = 0; j < 8; ++j) {
      int row = rbase + rl + r, col = colBase + j * 16 + cl;
      sT[row * D + col] = (_Float16)fmaxf(acc[j][r], 0.0f);
    }
  __syncthreads();

  // e += z @ W2 + b2 + mlp_t  (residual read from global, in-place write)
#pragma unroll
  for (int j = 0; j < 8; ++j) {
    int col = colBase + j * 16 + cl;
    float bv = b2[col] + mlpT[col];
#pragma unroll
    for (int r = 0; r < 8; ++r) acc[j][r] = bv;
  }
  gemm_pass(acc, sT + rbase * D, W2, sB2, tid, lane, colBase);
#pragma unroll
  for (int r = 0; r < 8; ++r)
#pragma unroll
    for (int j = 0; j < 8; ++j) {
      int row = rbase + rl + r, col = colBase + j * 16 + cl;
      size_t gi = (size_t)(r0 + row) * D + col;
      e16[gi] = (_Float16)((float)e16[gi] + acc[j][r]);
    }
}

// h += relu(LN(HU + agg) * s + b)
__global__ __launch_bounds__(256) void k_node_update(
    float* __restrict__ h32, _Float16* __restrict__ h16,
    const float* __restrict__ HU, const float* __restrict__ agg,
    const float* __restrict__ nnS, const float* __restrict__ nnB) {
  __shared__ float red[256], red2[256];
  int i = blockIdx.x, c = threadIdx.x;
  size_t gi = (size_t)i * D + c;
  float v = HU[gi] + agg[gi];
  red[c] = v; red2[c] = v * v;
  __syncthreads();
  for (int s = 128; s > 0; s >>= 1) {
    if (c < s) { red[c] += red[c + s]; red2[c] += red2[c + s]; }
    __syncthreads();
  }
  float mu   = red[0] * (1.0f / D);
  float var  = red2[0] * (1.0f / D) - mu * mu;
  float rstd = rsqrtf(var + 1e-5f);
  float x  = fmaxf((v - mu) * rstd * nnS[c] + nnB[c], 0.0f);
  float hn = h32[gi] + x;
  h32[gi] = hn;
  h16[gi] = (_Float16)hn;
}

// out = relu(e @ hw1 + hb1) @ hw2 + hb2   (WMMA + skinny VALU tail)
__global__ __launch_bounds__(256) void k_head(
    const _Float16* __restrict__ e16,
    const _Float16* __restrict__ W1, const float* __restrict__ b1,
    const float* __restrict__ W2f, const float* __restrict__ b2,
    float* __restrict__ out) {
  __shared__ alignas(32) _Float16 sT[64 * D];   // e tile, reused as z tile
  __shared__ alignas(32) _Float16 sB2[2 * 32 * D];
  int tid = threadIdx.x, lane = tid & 31, wave = tid >> 5;
  int r0 = blockIdx.x * 64;
  int rbase   = (wave & 3) * 16;
  int colBase = (wave >> 2) * 128;
  int rl = (lane >> 4) << 3;
  int cl = lane & 15;

  stage_tile64(sT, e16, r0, EE, tid, wave);
  __syncthreads();

  v8f acc[8];
#pragma unroll
  for (int j = 0; j < 8; ++j) {
    float bv = b1[colBase + j * 16 + cl];
#pragma unroll
    for (int r = 0; r < 8; ++r) acc[j][r] = bv;
  }
  gemm_pass(acc, sT + rbase * D, W1, sB2, tid, lane, colBase);
#pragma unroll
  for (int r = 0; r < 8; ++r)
#pragma unroll
    for (int j = 0; j < 8; ++j) {
      int row = rbase + rl + r, col = colBase + j * 16 + cl;
      sT[row * D + col] = (_Float16)fmaxf(acc[j][r], 0.0f);
    }
  __syncthreads();

  if (tid < 128) {
    int row = tid >> 1, cc = tid & 1;
    float s = b2[cc];
    for (int k = 0; k < D; ++k) s += (float)sT[row * D + k] * W2f[k * 2 + cc];
    out[(size_t)(r0 + row) * 2 + cc] = s;
  }
}

// ---------------------------------------------------------------------------
// Host driver
// ---------------------------------------------------------------------------
extern "C" void kernel_launch(void* const* d_in, const int* in_sizes, int n_in,
                              void* d_out, int out_size, void* d_ws, size_t ws_size,
                              hipStream_t stream) {
  (void)in_sizes; (void)n_in; (void)out_size; (void)ws_size;
  const float* coords = (const float*)d_in[0];
  const int*   eidx   = (const int*)d_in[1];
  const float* dist   = (const float*)d_in[2];
  const float* xt     = (const float*)d_in[3];
  const float* tt     = (const float*)d_in[4];
  const float* P   = (const float*)d_in[5];
  const float* Q   = (const float*)d_in[6];
  const float* R   = (const float*)d_in[7];
  const float* U   = (const float*)d_in[8];
  const float* V   = (const float*)d_in[9];
  const float* enS = (const float*)d_in[10];
  const float* enB = (const float*)d_in[11];
  const float* ew1 = (const float*)d_in[12];
  const float* eb1 = (const float*)d_in[13];
  const float* ew2 = (const float*)d_in[14];
  const float* eb2 = (const float*)d_in[15];
  const float* tw1 = (const float*)d_in[16];
  const float* tb1 = (const float*)d_in[17];
  const float* tw2 = (const float*)d_in[18];
  const float* tb2 = (const float*)d_in[19];
  const float* nnS = (const float*)d_in[20];
  const float* nnB = (const float*)d_in[21];
  const float* tpW1 = (const float*)d_in[22];
  const float* tpB1 = (const float*)d_in[23];
  const float* tpW2 = (const float*)d_in[24];
  const float* tpB2 = (const float*)d_in[25];
  const float* hW1 = (const float*)d_in[26];
  const float* hB1 = (const float*)d_in[27];
  const float* hW2 = (const float*)d_in[28];
  const float* hB2 = (const float*)d_in[29];

  const int* srcI = eidx;
  const int* dstI = eidx + EE;

  char* ws = (char*)d_ws;
  size_t off = 0;
  auto carve = [&](size_t bytes) -> char* {
    char* p = ws + off;
    off += (bytes + 255) & ~(size_t)255;
    return p;
  };
  float*    h32  = (float*)carve((size_t)NN * D * 4);
  _Float16* h16  = (_Float16*)carve((size_t)NN * D * 2);
  _Float16* e16  = (_Float16*)carve((size_t)EE * D * 2);
  _Float16* HQ   = (_Float16*)carve((size_t)NN * D * 2);
  _Float16* HR   = (_Float16*)carve((size_t)NN * D * 2);
  _Float16* HV   = (_Float16*)carve((size_t)NN * D * 2);
  float*    HU   = (float*)carve((size_t)NN * D * 4);
  float*    agg  = (float*)carve((size_t)NN * D * 4);
  float*    te   = (float*)carve(D * 4);
  float*    mlpt = (float*)carve(D * 4);
  const size_t WM = (size_t)LAY * D * D;
  _Float16* Pw  = (_Float16*)carve(WM * 2);
  _Float16* Qw  = (_Float16*)carve(WM * 2);
  _Float16* Rw  = (_Float16*)carve(WM * 2);
  _Float16* Uw  = (_Float16*)carve(WM * 2);
  _Float16* Vw  = (_Float16*)carve(WM * 2);
  _Float16* W1w = (_Float16*)carve(WM * 2);
  _Float16* W2w = (_Float16*)carve(WM * 2);
  _Float16* hW1w = (_Float16*)carve((size_t)D * D * 2);

  // f32 -> f16 weight conversion (each call; deterministic).
  const int cvtB = ((int)WM + 255) / 256;
  k_cvt<<<cvtB, 256, 0, stream>>>(P,   Pw,  (int)WM);
  k_cvt<<<cvtB, 256, 0, stream>>>(Q,   Qw,  (int)WM);
  k_cvt<<<cvtB, 256, 0, stream>>>(R,   Rw,  (int)WM);
  k_cvt<<<cvtB, 256, 0, stream>>>(U,   Uw,  (int)WM);
  k_cvt<<<cvtB, 256, 0, stream>>>(V,   Vw,  (int)WM);
  k_cvt<<<cvtB, 256, 0, stream>>>(ew1, W1w, (int)WM);
  k_cvt<<<cvtB, 256, 0, stream>>>(ew2, W2w, (int)WM);
  k_cvt<<<(D * D + 255) / 256, 256, 0, stream>>>(hW1, hW1w, D * D);

  // Embeddings + time projection.
  k_embed_nodes<<<(NN * D + 255) / 256, 256, 0, stream>>>(coords, h32, h16);
  {
    long long n = (long long)EE * D;
    k_embed_edges<<<(int)((n + 255) / 256), 256, 0, stream>>>(dist, xt, e16);
  }
  k_time_embed<<<1, 256, 0, stream>>>(tt, tpW1, tpB1, tpW2, tpB2, te);

  const int gemmBlocks = (NN + 63) / 64;
  const int edgeBlocks = EE / 64;

  for (int l = 0; l < LAY; ++l) {
    const size_t wOff = (size_t)l * D * D;
    const size_t vOff = (size_t)l * D;

    k_zero<<<(NN * D + 255) / 256, 256, 0, stream>>>(agg, NN * D);

    k_gemm256<<<gemmBlocks, 256, 0, stream>>>(h16, Qw + wOff, HQ, nullptr, NN);
    k_gemm256<<<gemmBlocks, 256, 0, stream>>>(h16, Rw + wOff, HR, nullptr, NN);
    k_gemm256<<<gemmBlocks, 256, 0, stream>>>(h16, Vw + wOff, HV, nullptr, NN);
    k_gemm256<<<gemmBlocks, 256, 0, stream>>>(h16, Uw + wOff, nullptr, HU, NN);

    k_mlp_t<<<1, 256, 0, stream>>>(te, tw1 + wOff, tb1 + vOff,
                                   tw2 + wOff, tb2 + vOff, mlpt);

    k_edge_layer<<<edgeBlocks, 256, 0, stream>>>(
        e16, srcI, dstI, HQ, HR, HV,
        Pw + wOff, W1w + wOff, W2w + wOff,
        enS + vOff, enB + vOff, eb1 + vOff, eb2 + vOff, mlpt, agg);

    k_node_update<<<NN, 256, 0, stream>>>(h32, h16, HU, agg,
                                          nnS + vOff, nnB + vOff);
  }

  k_head<<<edgeBlocks, 256, 0, stream>>>(e16, hW1w, hB1, hW2, hB2, (float*)d_out);
}
